// QuantizedLinear_29884382445885
// MI455X (gfx1250) — matrix operand order repaired and checked by reading, not compiled, not run on hardware
//
#include <hip/hip_runtime.h>

// out = x @ (W*scale).T + bias
//   x: [M=4096, K=4096] f32, W: [N=11008, K=4096] f32 (int8-valued),
//   scale: [N,1] f32, bias: [N] f32, out: [M, N] f32.
#define M_DIM 4096
#define N_DIM 11008
#define K_DIM 4096

#define BLK_M 128
#define BLK_N 128
#define BLK_K 32
#define NK    (K_DIM / BLK_K)
#define LDS_K 40  // padded row (halfs): 80B stride, breaks bank conflicts, keeps 16B align

typedef __attribute__((ext_vector_type(16))) _Float16 v16h;
typedef __attribute__((ext_vector_type(8)))  _Float16 v8h;
typedef __attribute__((ext_vector_type(4)))  _Float16 v4h;
typedef __attribute__((ext_vector_type(8)))  float    v8f;

union F16Frag { v16h v; v8h h[2]; };

__global__ __launch_bounds__(256, 2)
void qlinear_wmma_f16(const float* __restrict__ x,
                      const float* __restrict__ qw,
                      const float* __restrict__ scales,
                      const float* __restrict__ bias,
                      float* __restrict__ out) {
    // Double-buffered LDS tiles: 2 * (128*40 halfs) per matrix = 40 KB total.
    __shared__ __align__(16) _Float16 As[2][BLK_M * LDS_K];
    __shared__ __align__(16) _Float16 Bs[2][BLK_N * LDS_K];

    const int tid    = threadIdx.x;
    const int lane   = tid & 31;           // wave32
    const int wid    = tid >> 5;           // 8 waves
    const int lrow   = lane & 15;
    const int kgrp   = lane >> 4;          // 0 or 1 (low/high half of wave)
    const int wave_m = wid & 1;            // 2 waves along M
    const int wave_n = wid >> 1;           // 4 waves along N

    const int blkM = blockIdx.y * BLK_M;
    const int blkN = blockIdx.x * BLK_N;

    // Global staging: each thread loads one float4 per 32-row slab, 4 slabs
    // -> full 128x32 f32 tile per matrix. Fully coalesced global_load_b128.
    const int rowBase = tid >> 3;          // 0..31
    const int col4    = (tid & 7) * 4;     // 0,4,...,28

    const float* xg = x  + (size_t)blkM * K_DIM + col4;
    const float* wg = qw + (size_t)blkN * K_DIM + col4;

    v8f acc[4][2];
    const v8f vzero = {0.f, 0.f, 0.f, 0.f, 0.f, 0.f, 0.f, 0.f};
    #pragma unroll
    for (int i = 0; i < 4; ++i)
        #pragma unroll
        for (int j = 0; j < 2; ++j)
            acc[i][j] = vzero;

    float4 ra[4], rb[4];

    // ---- prologue: stage K-tile 0 into buffer 0 ----
    #pragma unroll
    for (int i = 0; i < 4; ++i) {
        const size_t roff = (size_t)(rowBase + i * 32) * K_DIM;
        ra[i] = *(const float4*)(xg + roff);
        rb[i] = *(const float4*)(wg + roff);
    }
    #pragma unroll
    for (int i = 0; i < 4; ++i) {
        const int r = rowBase + i * 32;
        v4h ha = {(_Float16)ra[i].x, (_Float16)ra[i].y, (_Float16)ra[i].z, (_Float16)ra[i].w};
        *(v4h*)(&As[0][r * LDS_K + col4]) = ha;
        v4h hb = {(_Float16)rb[i].x, (_Float16)rb[i].y, (_Float16)rb[i].z, (_Float16)rb[i].w};
        *(v4h*)(&Bs[0][r * LDS_K + col4]) = hb;
    }
    __syncthreads();

    for (int kt = 0; kt < NK; ++kt) {
        const int  cur     = kt & 1;
        const int  nxt     = cur ^ 1;
        const bool hasNext = (kt + 1) < NK;

        // (1) issue global loads for tile kt+1 (latency hidden under WMMA below)
        if (hasNext) {
            const int k1 = (kt + 1) * BLK_K;
            #pragma unroll
            for (int i = 0; i < 4; ++i) {
                const size_t roff = (size_t)(rowBase + i * 32) * K_DIM + k1;
                ra[i] = *(const float4*)(xg + roff);
                rb[i] = *(const float4*)(wg + roff);
            }
            // warm L2 for the tile after next (global_prefetch_b8)
            if (kt + 2 < NK) {
                __builtin_prefetch(xg + (size_t)rowBase * K_DIM + k1 + BLK_K, 0, 0);
                __builtin_prefetch(wg + (size_t)rowBase * K_DIM + k1 + BLK_K, 0, 0);
            }
        }

        // (2) compute on tile kt from LDS buffer `cur`
        // B fragment (kernel's own k order, not AMD's table order): lane holds column N=lane%16,
        //   K = h + 16*kgrp -> contiguous 16 halfs at [n][kgrp*16].
        F16Frag bf[2];
        #pragma unroll
        for (int tn = 0; tn < 2; ++tn) {
            const int n = wave_n * 32 + tn * 16 + lrow;
            const _Float16* p = &Bs[cur][n * LDS_K + kgrp * 16];
            bf[tn].h[0] = *(const v8h*)(p);
            bf[tn].h[1] = *(const v8h*)(p + 8);
        }
        // A layout (16-bit, 16x32): lane holds row M=lane%16;
    //   halfs 0..15 -> K = kgrp*16 + 0..15 (contiguous, matching B).
        #pragma unroll
        for (int tm = 0; tm < 4; ++tm) {
            const int m = wave_m * 64 + tm * 16 + lrow;
            const _Float16* p = &As[cur][m * LDS_K];
            F16Frag af;
            af.h[0] = *(const v8h*)(p + kgrp * 16);
            af.h[1] = *(const v8h*)(p + kgrp * 16 + 8);

            #pragma unroll
            for (int tn = 0; tn < 2; ++tn) {
                acc[tm][tn] = __builtin_amdgcn_wmma_f32_16x16x32_f16(
                    /*neg_a=*/false, af.v,
                    /*neg_b=*/false, bf[tn].v,
                    /*c_mod=*/(short)0, acc[tm][tn],
                    /*reuse_a=*/false, /*reuse_b=*/false);
            }
        }

        // (3) convert + store tile kt+1 into the other buffer
        if (hasNext) {
            #pragma unroll
            for (int i = 0; i < 4; ++i) {
                const int r = rowBase + i * 32;
                v4h ha = {(_Float16)ra[i].x, (_Float16)ra[i].y, (_Float16)ra[i].z, (_Float16)ra[i].w};
                *(v4h*)(&As[nxt][r * LDS_K + col4]) = ha;
                v4h hb = {(_Float16)rb[i].x, (_Float16)rb[i].y, (_Float16)rb[i].z, (_Float16)rb[i].w};
                *(v4h*)(&Bs[nxt][r * LDS_K + col4]) = hb;
            }
        }

        // (4) single barrier per K-step: fences reads of `cur` (overwritten at
        //     kt+2) and publishes writes of `nxt` for kt+1.
        __syncthreads();
    }

    // ---- epilogue: per-channel scale + bias, f32 stores ----
    // C/D layout: lane holds N=lane%16; VGPR r -> M = r + 8*kgrp.
    #pragma unroll
    for (int tn = 0; tn < 2; ++tn) {
        const int n  = blkN + wave_n * 32 + tn * 16 + lrow;
        const float s = scales[n];
        const float b = bias[n];
        #pragma unroll
        for (int tm = 0; tm < 4; ++tm) {
            const int mbase = blkM + wave_m * 64 + tm * 16 + kgrp * 8;
            #pragma unroll
            for (int r = 0; r < 8; ++r) {
                out[(size_t)(mbase + r) * N_DIM + n] = acc[tm][tn][r] * s + b;
            }
        }
    }
}

extern "C" void kernel_launch(void* const* d_in, const int* in_sizes, int n_in,
                              void* d_out, int out_size, void* d_ws, size_t ws_size,
                              hipStream_t stream) {
    const float* x      = (const float*)d_in[0];
    const float* qw     = (const float*)d_in[1];
    const float* scales = (const float*)d_in[2];
    const float* bias   = (const float*)d_in[3];
    float* out          = (float*)d_out;

    dim3 grid(N_DIM / BLK_N, M_DIM / BLK_M);  // (86, 32)
    dim3 block(256);
    hipLaunchKernelGGL(qlinear_wmma_f16, grid, block, 0, stream,
                       x, qw, scales, bias, out);
}
